// BayesianSphericalUNet_74191265071499
// MI455X (gfx1250) — matrix operand order repaired and checked by reading, not compiled, not run on hardware
//
#include <hip/hip_runtime.h>
#include <hip/hip_bf16.h>
#include <stdint.h>

// ---------------------------------------------------------------------------
// Types for CDNA5 WMMA (wave32)
// ---------------------------------------------------------------------------
typedef __attribute__((ext_vector_type(16))) __bf16       v16bf;
typedef __attribute__((ext_vector_type(8)))  float        v8f;
typedef __attribute__((ext_vector_type(8)))  unsigned int v8u;

#define GRID256(n) (((n) + 255) / 256)

// ---------------------------------------------------------------------------
// Helpers
// ---------------------------------------------------------------------------
__device__ __forceinline__ unsigned short f32_to_bf16(float f) {
    uint32_t u = __builtin_bit_cast(uint32_t, f);
    uint32_t r = (u + 0x7FFFu + ((u >> 16) & 1u)) >> 16;   // round-to-nearest-even
    return (unsigned short)r;
}

__device__ __forceinline__ float hash_uniform(uint32_t idx, uint32_t ch) {
    uint32_t s = idx * 0x9E3779B9u ^ ch * 0x85EBCA6Bu ^ 0x27220A95u;
    s ^= s >> 16; s *= 0x7FEB352Du;
    s ^= s >> 15; s *= 0x846CA68Bu;
    s ^= s >> 16;
    return (float)(s >> 8) * (1.0f / 16777216.0f);
}

// spatial concrete dropout mask value for (layer idx, channel)
__device__ __forceinline__ float cd_mask_val(float p_logit, int idx, int ch) {
    const float EPS = 1e-7f;
    float p = 1.0f / (1.0f + __expf(-p_logit));
    float u = hash_uniform((uint32_t)idx, (uint32_t)ch);
    // TEMP = 2/3 -> multiply by 1.5
    float lg = (__logf(p + EPS) - __logf(1.0f - p + EPS)
              + __logf(u + EPS) - __logf(1.0f - u + EPS)) * 1.5f;
    float drop = 1.0f / (1.0f + __expf(-lg));
    return (1.0f - drop) / (1.0f - p);
}

// ---------------------------------------------------------------------------
// 1) Concrete-dropout mask apply: xm[v,c] = x[v,c] * mask(c)
// ---------------------------------------------------------------------------
__global__ void apply_cd_mask(const float* __restrict__ x,
                              const float* __restrict__ p_logit,
                              float* __restrict__ xm,
                              int idx, int V, int C) {
    int gid = blockIdx.x * blockDim.x + threadIdx.x;
    if (gid >= V * C) return;
    int c = gid % C;
    xm[gid] = x[gid] * cd_mask_val(p_logit[0], idx, c);
}

// ---------------------------------------------------------------------------
// 2) Chebyshev terms. A is [V][KDp] bf16, KD = 3*Cin, KDp = KD rounded to 32.
//    pass1: A[:,0:C] = xm ; T1 = L*xm ; A[:,C:2C] = T1
//    pass2: A[:,2C:3C] = 2*L*T1 - xm
// ---------------------------------------------------------------------------
__global__ void cheb_pass1(const float* __restrict__ xm,
                           const int* __restrict__ cols,
                           const float* __restrict__ vals,
                           float* __restrict__ T1,
                           unsigned short* __restrict__ A,
                           int V, int C, int KDp) {
    int gid = blockIdx.x * blockDim.x + threadIdx.x;
    if (gid >= V * C) return;
    int v = gid / C, c = gid % C;
    float t = 0.0f;
#pragma unroll
    for (int nb = 0; nb < 9; ++nb) {
        int u = cols[v * 9 + nb];
        t = fmaf(vals[v * 9 + nb], xm[(size_t)u * C + c], t);
    }
    T1[gid] = t;
    unsigned short* Ar = A + (size_t)v * KDp;
    Ar[c]     = f32_to_bf16(xm[gid]);
    Ar[C + c] = f32_to_bf16(t);
}

__global__ void cheb_pass2(const float* __restrict__ xm,
                           const float* __restrict__ T1,
                           const int* __restrict__ cols,
                           const float* __restrict__ vals,
                           unsigned short* __restrict__ A,
                           int V, int C, int KDp) {
    int gid = blockIdx.x * blockDim.x + threadIdx.x;
    if (gid >= V * C) return;
    int v = gid / C, c = gid % C;
    float t = 0.0f;
#pragma unroll
    for (int nb = 0; nb < 9; ++nb) {
        int u = cols[v * 9 + nb];
        t = fmaf(vals[v * 9 + nb], T1[(size_t)u * C + c], t);
    }
    float t2 = 2.0f * t - xm[gid];
    A[(size_t)v * KDp + 2 * C + c] = f32_to_bf16(t2);
}

__global__ void pad_A(unsigned short* __restrict__ A, int V, int KD, int KDp) {
    int per = KDp - KD;
    int gid = blockIdx.x * blockDim.x + threadIdx.x;
    if (gid >= V * per) return;
    int v = gid / per, j = gid % per;
    A[(size_t)v * KDp + KD + j] = 0;
}

// ---------------------------------------------------------------------------
// 3) Weight conversion: W [K,Cin,Cout] f32 -> Wt [Cout][KDp] bf16 (transposed)
//    W[(k*Cin+ci)*Cout + o] -> Wt[o*KDp + (k*Cin+ci)]
// ---------------------------------------------------------------------------
__global__ void conv_weights(const float* __restrict__ W,
                             unsigned short* __restrict__ Wt,
                             int Cout, int KD, int KDp) {
    int gid = blockIdx.x * blockDim.x + threadIdx.x;
    if (gid >= Cout * KDp) return;
    int o = gid / KDp, kd = gid % KDp;
    float v = (kd < KD) ? W[(size_t)kd * Cout + o] : 0.0f;
    Wt[(size_t)o * KDp + kd] = f32_to_bf16(v);
}

// ---------------------------------------------------------------------------
// 4) WMMA GEMM: Y[v,o] = sum_kd A[v,kd]*Wt[o,kd] + bias[o]
//    Each wave computes a 16x64 output tile: one A fragment feeds 4 WMMAs per
//    32-deep k-step, so the streamed A matrix is read at most Cout/64 times
//    (exactly once for the two largest layers).
//    bf16 A fragment layout (16x32): lanes 0-15 hold K 0-7 & 16-23,
//    lanes 16-31 hold K 8-15 & 24-31, row m = lane&15.
//    bf16 B fragment (32x16): col n = lane&15, lanes 0-15 K 0-7 & 16-23,
//    lanes 16-31 K 8-15 & 24-31 (Wt is pre-transposed so these are contiguous).
// ---------------------------------------------------------------------------
__device__ __forceinline__ v16bf frag_from(const unsigned short* p0,
                                           const unsigned short* p1) {
    uint4 lo = *(const uint4*)p0;
    uint4 hi = *(const uint4*)p1;
    v8u u;
    u[0] = lo.x; u[1] = lo.y; u[2] = lo.z; u[3] = lo.w;
    u[4] = hi.x; u[5] = hi.y; u[6] = hi.z; u[7] = hi.w;
    return __builtin_bit_cast(v16bf, u);
}

__global__ __launch_bounds__(256) void cheb_gemm_wmma(
        const unsigned short* __restrict__ A,
        const unsigned short* __restrict__ Wt,
        const float* __restrict__ bias,
        float* __restrict__ Y,
        int V, int KDp, int Cout) {
    const int lane = threadIdx.x & 31;
    const int wave = threadIdx.x >> 5;
    const int tn4 = Cout >> 6;                 // 64-wide column tiles
    const int total = (V >> 4) * tn4;
    int tile = blockIdx.x * 8 + wave;
    if (tile >= total) return;                 // wave-uniform; EXEC stays all-ones
    const int tm = tile / tn4, tn = tile % tn4;
    const int half = lane >> 4;                // k-group within fragment
    const int l16  = lane & 15;
    const int m  = (tm << 4) + l16;
    const int n0 = (tn << 6) + l16;

    const unsigned short* Ar = A + (size_t)m * KDp + half * 8;
    const unsigned short* B[4];
    B[0] = Wt + (size_t)n0 * KDp + half * 8;
    B[1] = B[0] + (size_t)16 * KDp;
    B[2] = B[0] + (size_t)32 * KDp;
    B[3] = B[0] + (size_t)48 * KDp;

    v8f acc[4] = {{}, {}, {}, {}};
    for (int kk = 0; kk < KDp; kk += 32) {
        __builtin_prefetch((const void*)(Ar + kk + 128), 0, 3);  // near (WGP) scope
        v16bf a = frag_from(Ar + kk, Ar + kk + 16);
#pragma unroll
        for (int t = 0; t < 4; ++t) {
            v16bf b = frag_from(B[t] + kk, B[t] + kk + 16);
            acc[t] = __builtin_amdgcn_wmma_f32_16x16x32_bf16(
                         false, a, false, b, (short)0, acc[t], false, false);
        }
    }
    // C/D layout: VGPR r holds row r (lanes 0-15) / row r+8 (lanes 16-31)
    float* Yb = Y + (size_t)((tm << 4) + (half << 3)) * Cout + n0;
#pragma unroll
    for (int t = 0; t < 4; ++t) {
        const float bn = bias[n0 + 16 * t];
#pragma unroll
        for (int r = 0; r < 8; ++r)
            Yb[(size_t)r * Cout + 16 * t] = acc[t][r] + bn;
    }
}

// ---------------------------------------------------------------------------
// 5) BatchNorm (population stats over V) + ReLU
// ---------------------------------------------------------------------------
__global__ __launch_bounds__(256) void bn_stats(const float* __restrict__ Y,
                                                int V, int C,
                                                float* __restrict__ mean,
                                                float* __restrict__ invstd) {
    int c = blockIdx.x;
    float s = 0.0f, s2 = 0.0f;
    for (int v = threadIdx.x; v < V; v += 256) {
        float y = Y[(size_t)v * C + c];
        s += y; s2 += y * y;
    }
    __shared__ float sh[512];
    sh[threadIdx.x]       = s;
    sh[256 + threadIdx.x] = s2;
    __syncthreads();
    for (int off = 128; off; off >>= 1) {
        if (threadIdx.x < off) {
            sh[threadIdx.x]       += sh[threadIdx.x + off];
            sh[256 + threadIdx.x] += sh[256 + threadIdx.x + off];
        }
        __syncthreads();
    }
    if (threadIdx.x == 0) {
        float m   = sh[0] / (float)V;
        float var = sh[256] / (float)V - m * m;
        mean[c]   = m;
        invstd[c] = rsqrtf(var + 1e-5f);
    }
}

__global__ void bn_apply_relu(float* __restrict__ Y, int V, int C,
                              const float* __restrict__ mean,
                              const float* __restrict__ invstd,
                              const float* __restrict__ g,
                              const float* __restrict__ b) {
    int gid = blockIdx.x * blockDim.x + threadIdx.x;
    if (gid >= V * C) return;
    int c = gid % C;
    float y = (Y[gid] - mean[c]) * invstd[c] * g[c] + b[c];
    Y[gid] = fmaxf(y, 0.0f);
}

// ---------------------------------------------------------------------------
// 6) HealpixDownsample(4) = max-pool over groups of 4 vertices
// ---------------------------------------------------------------------------
__global__ void maxpool4(const float* __restrict__ in, float* __restrict__ out,
                         int Vout, int C) {
    int gid = blockIdx.x * blockDim.x + threadIdx.x;
    if (gid >= Vout * C) return;
    int v = gid / C, c = gid % C;
    float m = in[(size_t)(4 * v) * C + c];
#pragma unroll
    for (int j = 1; j < 4; ++j)
        m = fmaxf(m, in[(size_t)(4 * v + j) * C + c]);
    out[gid] = m;
}

// ---------------------------------------------------------------------------
// 7) Upsample(nearest, x4) + skip concat along channels
// ---------------------------------------------------------------------------
__global__ void upsample_concat(const float* __restrict__ up,
                                const float* __restrict__ skip,
                                float* __restrict__ out,
                                int Vout, int Cu, int Cs) {
    int Ct = Cu + Cs;
    int gid = blockIdx.x * blockDim.x + threadIdx.x;
    if (gid >= Vout * Ct) return;
    int v = gid / Ct, c = gid % Ct;
    out[gid] = (c < Cu) ? up[(size_t)(v >> 2) * Cu + c]
                        : skip[(size_t)v * Cs + (c - Cu)];
}

// ---------------------------------------------------------------------------
// 8) Final heads: mu/logvar = masked 1x64 projections (channels-first mask)
// ---------------------------------------------------------------------------
__global__ void heads_kernel(const float* __restrict__ x,   // [V][64]
                             const float* __restrict__ muW,
                             const float* __restrict__ mub,
                             const float* __restrict__ mup,
                             const float* __restrict__ lvW,
                             const float* __restrict__ lvb,
                             const float* __restrict__ lvp,
                             float* __restrict__ out,
                             int V, int idx_mu, int idx_lv) {
    int v = blockIdx.x * blockDim.x + threadIdx.x;
    if (v >= V) return;
    float pm = mup[0], pl = lvp[0];
    float smu = 0.0f, slv = 0.0f;
#pragma unroll 4
    for (int c = 0; c < 64; ++c) {
        float xv = x[(size_t)v * 64 + c];
        smu = fmaf(xv * cd_mask_val(pm, idx_mu, c), muW[c], smu);
        slv = fmaf(xv * cd_mask_val(pl, idx_lv, c), lvW[c], slv);
    }
    out[v]     = smu + mub[0];
    out[V + v] = slv + lvb[0];
}

// ---------------------------------------------------------------------------
// Host orchestration
// ---------------------------------------------------------------------------
namespace {

struct Scratch {
    float* xm;            // masked input           [Vmax*Cmax]
    float* t1;            // T1 Laplacian term      [Vmax*Cmax]
    unsigned short* A;    // Cheb A matrix bf16     [Vmax*KDpmax]
    unsigned short* Wt;   // bf16 transposed weights
    float* y1;            // mid activation
    float* mean;
    float* invstd;
};

// block leaf order (JAX sorted keys): b1,b2,c1.W,c1.b,c1.p,c2.W,c2.b,c2.p,g1,g2
enum { L_B1 = 0, L_B2, L_C1W, L_C1B, L_C1P, L_C2W, L_C2B, L_C2P, L_G1, L_G2 };

void run_cheb(hipStream_t s, const Scratch& ws,
              const float* xin, int V, int Cin, int Cout,
              const int* cols, const float* vals,
              const float* W, const float* bias, const float* plogit,
              int maskIdx, float* Y) {
    int KD = 3 * Cin;
    int KDp = (KD + 31) & ~31;
    int n = V * Cin;
    apply_cd_mask<<<GRID256(n), 256, 0, s>>>(xin, plogit, ws.xm, maskIdx, V, Cin);
    cheb_pass1<<<GRID256(n), 256, 0, s>>>(ws.xm, cols, vals, ws.t1, ws.A, V, Cin, KDp);
    cheb_pass2<<<GRID256(n), 256, 0, s>>>(ws.xm, ws.t1, cols, vals, ws.A, V, Cin, KDp);
    if (KDp > KD) {
        int np = V * (KDp - KD);
        pad_A<<<GRID256(np), 256, 0, s>>>(ws.A, V, KD, KDp);
    }
    int nw = Cout * KDp;
    conv_weights<<<GRID256(nw), 256, 0, s>>>(W, ws.Wt, Cout, KD, KDp);
    int tiles = (V / 16) * (Cout / 64);
    cheb_gemm_wmma<<<(tiles + 7) / 8, 256, 0, s>>>(ws.A, ws.Wt, bias, Y, V, KDp, Cout);
}

void run_bn_relu(hipStream_t s, const Scratch& ws, float* Y, int V, int C,
                 const float* g, const float* b) {
    bn_stats<<<C, 256, 0, s>>>(Y, V, C, ws.mean, ws.invstd);
    bn_apply_relu<<<GRID256(V * C), 256, 0, s>>>(Y, V, C, ws.mean, ws.invstd, g, b);
}

void run_block(hipStream_t s, const Scratch& ws,
               const float* x, int V, int Cin, int Cmid, int Cout,
               const int* cols, const float* vals,
               void* const* L, int idx, float* out) {
    run_cheb(s, ws, x, V, Cin, Cmid, cols, vals,
             (const float*)L[L_C1W], (const float*)L[L_C1B],
             (const float*)L[L_C1P], idx, ws.y1);
    run_bn_relu(s, ws, ws.y1, V, Cmid, (const float*)L[L_G1], (const float*)L[L_B1]);
    run_cheb(s, ws, ws.y1, V, Cmid, Cout, cols, vals,
             (const float*)L[L_C2W], (const float*)L[L_C2B],
             (const float*)L[L_C2P], idx + 1, out);
    run_bn_relu(s, ws, out, V, Cout, (const float*)L[L_G2], (const float*)L[L_B2]);
}

} // namespace

extern "C" void kernel_launch(void* const* d_in, const int* in_sizes, int n_in,
                              void* d_out, int out_size, void* d_ws, size_t ws_size,
                              hipStream_t stream) {
    (void)in_sizes; (void)n_in; (void)out_size; (void)ws_size;

    const float* x = (const float*)d_in[0];
    const int*   cols[4] = { (const int*)d_in[1], (const int*)d_in[3],
                             (const int*)d_in[5], (const int*)d_in[7] };
    const float* vals[4] = { (const float*)d_in[2], (const float*)d_in[4],
                             (const float*)d_in[6], (const float*)d_in[8] };
    // params flattened in JAX sorted-key pytree order:
    // bott(10), dec0(10), dec1(10), dec2(10), enc0(10), enc1(10), enc2(10),
    // lv_W, lv_b, lv_p, mu_W, mu_b, mu_p
    void* const* P = d_in + 9;
    void* const* bott = P + 0;
    void* const* dec0 = P + 10;
    void* const* dec1 = P + 20;
    void* const* dec2 = P + 30;
    void* const* enc0 = P + 40;
    void* const* enc1 = P + 50;
    void* const* enc2 = P + 60;
    const float* lvW = (const float*)P[70];
    const float* lvb = (const float*)P[71];
    const float* lvp = (const float*)P[72];
    const float* muW = (const float*)P[73];
    const float* mub = (const float*)P[74];
    const float* mup = (const float*)P[75];

    // ---- workspace bump allocator ----
    char* base = (char*)d_ws;
    size_t off = 0;
    auto alloc = [&](size_t bytes) -> void* {
        void* p = base + off;
        off = (off + bytes + 255) & ~(size_t)255;
        return p;
    };
    Scratch ws;
    ws.xm     = (float*)alloc((size_t)49152 * 128 * sizeof(float));
    ws.t1     = (float*)alloc((size_t)49152 * 128 * sizeof(float));
    ws.A      = (unsigned short*)alloc((size_t)49152 * 384 * sizeof(unsigned short));
    ws.Wt     = (unsigned short*)alloc((size_t)1536 * 512 * sizeof(unsigned short));
    ws.y1     = (float*)alloc((size_t)49152 * 64 * sizeof(float));
    ws.mean   = (float*)alloc(512 * sizeof(float));
    ws.invstd = (float*)alloc(512 * sizeof(float));
    float* S0   = (float*)alloc((size_t)49152 * 64 * sizeof(float));   // enc0 skip
    float* S1   = (float*)alloc((size_t)12288 * 128 * sizeof(float));  // enc1 skip
    float* S2   = (float*)alloc((size_t)3072 * 256 * sizeof(float));   // enc2 skip
    float* curA = (float*)alloc((size_t)49152 * 128 * sizeof(float));
    float* curB = (float*)alloc((size_t)49152 * 128 * sizeof(float));

    // ---- encoder ----
    // enc0: V=49152, (9,64,64), lap 3 (finest)
    run_block(stream, ws, x, 49152, 9, 64, 64, cols[3], vals[3], enc0, 0, S0);
    maxpool4<<<GRID256(12288 * 64), 256, 0, stream>>>(S0, curA, 12288, 64);
    // enc1: V=12288, (64,128,128), lap 2
    run_block(stream, ws, curA, 12288, 64, 128, 128, cols[2], vals[2], enc1, 2, S1);
    maxpool4<<<GRID256(3072 * 128), 256, 0, stream>>>(S1, curA, 3072, 128);
    // enc2: V=3072, (128,256,256), lap 1
    run_block(stream, ws, curA, 3072, 128, 256, 256, cols[1], vals[1], enc2, 4, S2);
    maxpool4<<<GRID256(768 * 256), 256, 0, stream>>>(S2, curA, 768, 256);

    // ---- bottleneck: V=768, (256,512,256), lap 0 ----
    run_block(stream, ws, curA, 768, 256, 512, 256, cols[0], vals[0], bott, 6, curB);

    // ---- decoder ----
    // dec0: up 768->3072, concat S2 (256+256=512), (512,256,128), lap 1
    upsample_concat<<<GRID256(3072 * 512), 256, 0, stream>>>(curB, S2, curA, 3072, 256, 256);
    run_block(stream, ws, curA, 3072, 512, 256, 128, cols[1], vals[1], dec0, 8, curB);
    // dec1: up 3072->12288, concat S1 (128+128=256), (256,128,64), lap 2
    upsample_concat<<<GRID256(12288 * 256), 256, 0, stream>>>(curB, S1, curA, 12288, 128, 128);
    run_block(stream, ws, curA, 12288, 256, 128, 64, cols[2], vals[2], dec1, 10, curB);
    // dec2: up 12288->49152, concat S0 (64+64=128), (128,64,64), lap 3
    upsample_concat<<<GRID256(49152 * 128), 256, 0, stream>>>(curB, S0, curA, 49152, 64, 64);
    run_block(stream, ws, curA, 49152, 128, 64, 64, cols[3], vals[3], dec2, 12, curB);

    // ---- heads: mu / logvar over 64 channels, per-channel masks idx 14/15 ----
    heads_kernel<<<GRID256(49152), 256, 0, stream>>>(
        curB, muW, mub, mup, lvW, lvb, lvp, (float*)d_out, 49152, 14, 15);
}
